// FCAutoEncoder_65755949302193
// MI455X (gfx1250) — matrix operand order repaired and checked by reading, not compiled, not run on hardware
//
#include <hip/hip_runtime.h>

typedef __attribute__((ext_vector_type(16))) _Float16 v16h;
typedef __attribute__((ext_vector_type(8)))  _Float16 v8h;
typedef __attribute__((ext_vector_type(8)))  float    v8f;

#define BATCH   16384
#define BASE_D  1008
#define KP_BASE 1024
#define NP_BASE 1024
#define NBUCKET 5

// ---------------------------------------------------------------------------
__global__ void zero_counts_kernel(int* __restrict__ counts) {
    if (threadIdx.x < NBUCKET) counts[threadIdx.x] = 0;
}

// f32 [N,K] -> f16 [Npad,Kp], zero-padded both dims. Kp is a power of two.
__global__ void convert_pad_kernel(const float* __restrict__ src, _Float16* __restrict__ dst,
                                   int N, int K, int Npad, int Kp) {
    size_t idx = (size_t)blockIdx.x * blockDim.x + threadIdx.x;
    size_t total = (size_t)Npad * (size_t)Kp;
    if (idx >= total) return;
    const int ks = __builtin_ctz(Kp);
    int n  = (int)(idx >> ks);
    int kk = (int)(idx & (size_t)(Kp - 1));
    float v = (n < N && kk < K) ? src[(size_t)n * K + kk] : 0.f;
    dst[idx] = (_Float16)v;
}

// Per-row: bucket assignment (atomic compaction; final output independent of
// intra-bucket order), masked f16 conversion of x, zero K-pad cols of exph/dech.
__global__ void prepare_kernel(const float* __restrict__ x, const int* __restrict__ seqlen,
                               _Float16* __restrict__ xh, _Float16* __restrict__ exph,
                               _Float16* __restrict__ dech,
                               int* __restrict__ rowmap, int* __restrict__ counts) {
    const int row = blockIdx.x;
    const int L = seqlen[row];
    if (threadIdx.x == 0) {
        int k = (L == 36) ? 0 : (L == 72) ? 1 : (L == 144) ? 2 : (L == 288) ? 3 : 4;
        int pos = atomicAdd(&counts[k], 1);
        rowmap[k * BATCH + pos] = row;
    }
    const size_t rb = (size_t)row * KP_BASE;
    for (int j = threadIdx.x; j < KP_BASE; j += blockDim.x) {
        float v = (j < L) ? x[(size_t)row * BASE_D + j] : 0.f;   // L <= 1008 <= KP_BASE
        xh[rb + j] = (_Float16)v;
        if (j >= BASE_D) { exph[rb + j] = (_Float16)0.f; dech[rb + j] = (_Float16)0.f; }
    }
}

// ---------------------------------------------------------------------------
// Per-wave 32x64 output strip: 2 M-subtiles x 4 N-subtiles = 8 independent
// accumulators. Per K-step: 4x16B A loads + 4x32B B loads feed 8 WMMAs
// (21.3 f16-FLOP/byte), each B reused twice, chains 8 ops apart.
// A (16-bit 16x32): lane l = row (l&15); two 16B chunks at K = h*16, h*16+8.
// B (32x16):        lane l = col (l&15); one 32B chunk at K = h*16.
// C/D f32:          lane l -> n = l&15, m = vgpr + 8*h.   (h = l>>4)
__device__ __forceinline__ v16h load_a(const _Float16* __restrict__ arow, int k, int half) {
    v8h a0 = *(const v8h*)(arow + k + half * 16);
    v8h a1 = *(const v8h*)(arow + k + half * 16 + 8);
    v16h a;
#pragma unroll
    for (int e = 0; e < 8; ++e) { a[e] = a0[e]; a[e + 8] = a1[e]; }
    return a;
}

__device__ __forceinline__ void wmma_k_loop_2x4(const _Float16* __restrict__ arow0,
                                                const _Float16* __restrict__ arow1,
                                                const _Float16* __restrict__ wrow,
                                                size_t wsub,   // 16*Kp between N-subtiles
                                                int Kp, int half, v8f c[2][4]) {
    for (int k = 0; k < Kp; k += 32) {
        v16h a0 = load_a(arow0, k, half);
        v16h a1 = load_a(arow1, k, half);
#pragma unroll
        for (int j = 0; j < 4; ++j) {
            v16h b = *(const v16h*)(wrow + (size_t)j * wsub + k + half * 16);
            c[0][j] = __builtin_amdgcn_wmma_f32_16x16x32_f16(false, a0, false, b,
                                                             (short)0, c[0][j], false, false);
            c[1][j] = __builtin_amdgcn_wmma_f32_16x16x32_f16(false, a1, false, b,
                                                             (short)0, c[1][j], false, false);
        }
    }
}

// Dense: O[M,Np] = act(A[M,Kp] @ W[Npad,Kp]^T + bias), W zero-row padded.
__global__ __launch_bounds__(256)
void gemm_f16_kernel(const _Float16* __restrict__ A, const _Float16* __restrict__ W,
                     const float* __restrict__ bias,
                     _Float16* __restrict__ Oh, float* __restrict__ Of,
                     int N, int Npad, int Kp, int Np, int relu) {
    const int lane = threadIdx.x & 31;
    const int wave = threadIdx.x >> 5;
    const int half = lane >> 4;
    const int l15  = lane & 15;
    const int mtile0 = blockIdx.x * 32;
    const int ntile0 = (blockIdx.y * (blockDim.x >> 5) + wave) * 64;
    if (ntile0 >= Npad) return;

    const _Float16* arow0 = A + (size_t)(mtile0 + l15) * Kp;
    const _Float16* arow1 = arow0 + (size_t)16 * Kp;
    const _Float16* wrow  = W + (size_t)(ntile0 + l15) * Kp;
    v8f c[2][4] = {};
    wmma_k_loop_2x4(arow0, arow1, wrow, (size_t)16 * Kp, Kp, half, c);

#pragma unroll
    for (int j = 0; j < 4; ++j) {
        const int n = ntile0 + j * 16 + l15;
        if (n >= N) continue;
        const float bn = bias[n];
#pragma unroll
        for (int s = 0; s < 2; ++s) {
#pragma unroll
            for (int v = 0; v < 8; ++v) {
                float val = c[s][j][v] + bn;
                if (relu) val = fmaxf(val, 0.f);
                const int m = mtile0 + s * 16 + half * 8 + v;
                if (Oh) Oh[(size_t)m * Np + n] = (_Float16)val;
                else    Of[(size_t)m * Np + n] = val;
            }
        }
    }
}

// Bucketed: rows gathered/scattered through rowmap[bucket]; weight/bias picked
// by bucket. D rows depend only on their own A rows, so over-count lanes ride
// on row 0 and just skip their stores.
__global__ __launch_bounds__(256)
void gemm_bucket_kernel(const _Float16* __restrict__ A, const _Float16* __restrict__ Wall,
                        const float* __restrict__ biasAll,
                        const int* __restrict__ rowmap, const int* __restrict__ counts,
                        _Float16* __restrict__ Oh, float* __restrict__ Of,
                        int N, int Npad, int Kp, int Np) {
    const int bucket = blockIdx.z;
    const int cnt = counts[bucket];
    const int mtile0 = blockIdx.x * 32;
    if (mtile0 >= cnt) return;
    const int lane = threadIdx.x & 31;
    const int wave = threadIdx.x >> 5;
    const int half = lane >> 4;
    const int l15  = lane & 15;
    const int ntile0 = (blockIdx.y * (blockDim.x >> 5) + wave) * 64;
    if (ntile0 >= Npad) return;

    const int* rmap = rowmap + bucket * BATCH;
    const int mIdx0 = mtile0 + l15;
    const int mIdx1 = mtile0 + 16 + l15;
    const int grow0 = (mIdx0 < cnt) ? rmap[mIdx0] : 0;
    const int grow1 = (mIdx1 < cnt) ? rmap[mIdx1] : 0;
    const _Float16* arow0 = A + (size_t)grow0 * Kp;
    const _Float16* arow1 = A + (size_t)grow1 * Kp;
    const _Float16* wrow  = Wall + (size_t)bucket * Npad * Kp + (size_t)(ntile0 + l15) * Kp;
    v8f c[2][4] = {};
    wmma_k_loop_2x4(arow0, arow1, wrow, (size_t)16 * Kp, Kp, half, c);

#pragma unroll
    for (int j = 0; j < 4; ++j) {
        const int n = ntile0 + j * 16 + l15;
        if (n >= N) continue;
        const float bn = biasAll[bucket * BASE_D + n];
#pragma unroll
        for (int s = 0; s < 2; ++s) {
#pragma unroll
            for (int v = 0; v < 8; ++v) {
                const int mloc = mtile0 + s * 16 + half * 8 + v;
                if (mloc >= cnt) continue;
                const int orow = rmap[mloc];
                const float val = c[s][j][v] + bn;
                if (Oh) Oh[(size_t)orow * Np + n] = (_Float16)val;
                else    Of[(size_t)orow * Np + n] = val;
            }
        }
    }
}

// ---------------------------------------------------------------------------
extern "C" void kernel_launch(void* const* d_in, const int* in_sizes, int n_in,
                              void* d_out, int out_size, void* d_ws, size_t ws_size,
                              hipStream_t stream) {
    const float* x    = (const float*)d_in[0];
    const int* seqlen = (const int*)d_in[1];
    const float* Win  = (const float*)d_in[2];
    const float* bin_ = (const float*)d_in[3];
    const float* Wout = (const float*)d_in[4];
    const float* bout = (const float*)d_in[5];
    const float* We1  = (const float*)d_in[6];
    const float* be1  = (const float*)d_in[7];
    const float* We2  = (const float*)d_in[8];
    const float* be2  = (const float*)d_in[9];
    const float* We3  = (const float*)d_in[10];
    const float* be3  = (const float*)d_in[11];
    const float* Wd1  = (const float*)d_in[12];
    const float* bd1  = (const float*)d_in[13];
    const float* Wd2  = (const float*)d_in[14];
    const float* bd2  = (const float*)d_in[15];
    const float* Wd3  = (const float*)d_in[16];
    const float* bd3  = (const float*)d_in[17];
    float* out = (float*)d_out;

    char* ws = (char*)d_ws;
    size_t off = 0;
    auto alloc = [&](size_t bytes) -> char* {
        char* p = ws + off; off += (bytes + 255) & ~(size_t)255; return p;
    };
    int*      counts = (int*)alloc(NBUCKET * sizeof(int));
    int*      rowmap = (int*)alloc((size_t)NBUCKET * BATCH * sizeof(int));
    _Float16* xh     = (_Float16*)alloc((size_t)BATCH * KP_BASE * 2);
    _Float16* exph   = (_Float16*)alloc((size_t)BATCH * KP_BASE * 2);
    _Float16* h1     = (_Float16*)alloc((size_t)BATCH * 512 * 2);
    _Float16* h2     = (_Float16*)alloc((size_t)BATCH * 256 * 2);
    _Float16* lat    = (_Float16*)alloc((size_t)BATCH * 128 * 2);
    _Float16* h3     = (_Float16*)alloc((size_t)BATCH * 256 * 2);
    _Float16* h4     = (_Float16*)alloc((size_t)BATCH * 512 * 2);
    _Float16* dech   = (_Float16*)alloc((size_t)BATCH * KP_BASE * 2);
    _Float16* WinH   = (_Float16*)alloc((size_t)NBUCKET * NP_BASE * KP_BASE * 2);
    _Float16* WoutH  = (_Float16*)alloc((size_t)NBUCKET * NP_BASE * KP_BASE * 2);
    _Float16* We1H   = (_Float16*)alloc((size_t)512 * KP_BASE * 2);
    _Float16* We2H   = (_Float16*)alloc((size_t)256 * 512 * 2);
    _Float16* We3H   = (_Float16*)alloc((size_t)128 * 256 * 2);
    _Float16* Wd1H   = (_Float16*)alloc((size_t)256 * 128 * 2);
    _Float16* Wd2H   = (_Float16*)alloc((size_t)512 * 256 * 2);
    _Float16* Wd3H   = (_Float16*)alloc((size_t)NP_BASE * 512 * 2);

    hipLaunchKernelGGL(zero_counts_kernel, dim3(1), dim3(32), 0, stream, counts);

    auto conv = [&](const float* src, _Float16* dst, int N, int K, int Npad, int Kp) {
        size_t total = (size_t)Npad * Kp;
        hipLaunchKernelGGL(convert_pad_kernel, dim3((unsigned)((total + 255) / 256)), dim3(256),
                           0, stream, src, dst, N, K, Npad, Kp);
    };
    for (int k = 0; k < NBUCKET; ++k) {
        conv(Win  + (size_t)k * BASE_D * BASE_D, WinH  + (size_t)k * NP_BASE * KP_BASE,
             BASE_D, BASE_D, NP_BASE, KP_BASE);
        conv(Wout + (size_t)k * BASE_D * BASE_D, WoutH + (size_t)k * NP_BASE * KP_BASE,
             BASE_D, BASE_D, NP_BASE, KP_BASE);
    }
    conv(We1, We1H, 512,    BASE_D, 512,     KP_BASE);
    conv(We2, We2H, 256,    512,    256,     512);
    conv(We3, We3H, 128,    256,    128,     256);
    conv(Wd1, Wd1H, 256,    128,    256,     128);
    conv(Wd2, Wd2H, 512,    256,    512,     256);
    conv(Wd3, Wd3H, BASE_D, 512,    NP_BASE, 512);

    hipLaunchKernelGGL(prepare_kernel, dim3(BATCH), dim3(256), 0, stream,
                       x, seqlen, xh, exph, dech, rowmap, counts);

    // expand: per-bucket  exph = xh @ Win[k]^T + bin[k]
    hipLaunchKernelGGL(gemm_bucket_kernel, dim3(BATCH / 32, 2, NBUCKET), dim3(256), 0, stream,
                       xh, WinH, bin_, rowmap, counts, exph, (float*)nullptr,
                       BASE_D, NP_BASE, KP_BASE, KP_BASE);

    auto gemm = [&](const _Float16* A, const _Float16* W, const float* b,
                    _Float16* Oh, float* Of, int N, int Npad, int Kp, int Np, int relu) {
        int ygroups = ((Npad / 64) + 7) / 8;
        hipLaunchKernelGGL(gemm_f16_kernel, dim3(BATCH / 32, ygroups), dim3(256), 0, stream,
                           A, W, b, Oh, Of, N, Npad, Kp, Np, relu);
    };
    gemm(exph, We1H, be1, h1,   nullptr, 512,    512,     KP_BASE, 512,     1);
    gemm(h1,   We2H, be2, h2,   nullptr, 256,    256,     512,     256,     1);
    gemm(h2,   We3H, be3, lat,  nullptr, 128,    128,     256,     128,     0);
    gemm(lat,  Wd1H, bd1, h3,   nullptr, 256,    256,     128,     256,     1);
    gemm(h3,   Wd2H, bd2, h4,   nullptr, 512,    512,     256,     512,     1);
    gemm(h4,   Wd3H, bd3, dech, nullptr, BASE_D, NP_BASE, 512,     KP_BASE, 0);

    // contract: per-bucket  out = dech @ Wout[k]^T + bout[k]
    hipLaunchKernelGGL(gemm_bucket_kernel, dim3(BATCH / 32, 2, NBUCKET), dim3(256), 0, stream,
                       dech, WoutH, bout, rowmap, counts, (_Float16*)nullptr, out,
                       BASE_D, NP_BASE, KP_BASE, BASE_D);
}
